// RnnSenderFixedLengthGS_61203283968874
// MI455X (gfx1250) — matrix operand order repaired and checked by reading, not compiled, not run on hardware
//
#include <hip/hip_runtime.h>

// ---- problem constants (from reference) ------------------------------------
#define B_   128
#define IN_  512
#define E_   512
#define H_   512
#define V_   32000
#define T_   16
#define H3_  1536   // 3*H
#define KSPLIT 20   // K-split factor for the embedding GEMM (32000/20=1600, %32==0)

#ifndef __has_builtin
#define __has_builtin(x) 0
#endif
#if __has_builtin(__builtin_amdgcn_global_load_async_to_lds_b128)
#define HAVE_ASYNC_LDS 1
#else
#define HAVE_ASYNC_LDS 0
#endif

// ---- types for CDNA5 WMMA ---------------------------------------------------
typedef __attribute__((ext_vector_type(16))) __bf16 v16bf;
typedef __attribute__((ext_vector_type(8)))  float  v8f;
typedef __attribute__((ext_vector_type(4)))  int    v4i;

typedef __attribute__((address_space(1))) v4i* gv4i_p;  // global int4 vector ptr
typedef __attribute__((address_space(3))) v4i* lv4i_p;  // LDS int4 vector ptr

union FragBF {
  v16bf v;
  unsigned short u[16];
  uint4 q[2];
};

__device__ __forceinline__ unsigned short f2b(float f) {
  // fp32 -> bf16, round-to-nearest-even
  unsigned int x = __float_as_uint(f);
  x += 0x7FFFu + ((x >> 16) & 1u);
  return (unsigned short)(x >> 16);
}

// ---- generic WMMA GEMM ------------------------------------------------------
// C[128,N] = A_bf16[128,K] * B[N,K]^T (+bias[N]) (+extra[128,N])
// PREBF=true : B is bf16 (pre-converted weights in workspace) -> pure-load path.
// PREBF=false: B is fp32 (original weights), converted in-register.
// Block = 256 threads = 8 wave32s. Each wave owns a 16-wide N strip and the
// full M=128 (8 accumulators), so one B fragment feeds 8 WMMAs per K-chunk.
// The shared 128x32 A tile is staged in LDS in fragment order per K-chunk,
// using GLOBAL_LOAD_ASYNC_TO_LDS (ASYNCcnt) when the toolchain exposes it.
// gridDim.z splits K: k_begin = blockIdx.z*k_len, C += blockIdx.z*zstride.
template <bool PREBF>
__global__ __launch_bounds__(256)
void gemm_core(const unsigned short* __restrict__ A, int lda,
               const void* __restrict__ Bw, int ldb,
               const float* __restrict__ bias,
               const float* __restrict__ extra, int lde,
               float* __restrict__ C, long ldc, long zstride,
               unsigned short* __restrict__ Cbf, int ldcb,
               int k_len)
{
  __shared__ uint4 ldsA[512];  // 128 rows x 32 K of bf16, fragment order (8 KB)

  const int lane = threadIdx.x & 31;
  const int wave = threadIdx.x >> 5;
  const int col0 = (blockIdx.x << 7) + (wave << 4);  // N strip for this wave
  const int mr   = lane & 15;
  const int hi   = lane >> 4;
  const int kb   = blockIdx.z * k_len;
  C += (long)blockIdx.z * zstride;

  const unsigned short* bb = nullptr;
  const float*          bf = nullptr;
  if constexpr (PREBF) bb = (const unsigned short*)Bw + (size_t)(col0 + mr) * ldb;
  else                 bf = (const float*)Bw          + (size_t)(col0 + mr) * ldb;

  v8f acc[8];
#pragma unroll
  for (int mt = 0; mt < 8; ++mt) acc[mt] = v8f{};

  for (int k0 = kb; k0 < kb + k_len; k0 += 32) {
    // ---- cooperative A-tile stage (global -> LDS, fragment order) ----------
    __syncthreads();  // previous iteration's LDS reads done
#pragma unroll
    for (int s = 0; s < 2; ++s) {
      const int id = threadIdx.x * 2 + s;       // 0..511 : (row, k-octet)
      const int r  = id >> 2;
      const int j  = id & 3;
      const unsigned short* src = A + (size_t)r * lda + k0 + j * 8;
      const int mt = r >> 4, rr = r & 15;
      const int ln = rr + ((j >> 1) << 4);
      const int ch = j & 1;
      uint4* dst = &ldsA[(mt * 32 + ln) * 2 + ch];
#if HAVE_ASYNC_LDS
      __builtin_amdgcn_global_load_async_to_lds_b128(
          (gv4i_p)(unsigned short*)src, (lv4i_p)dst, 0, 0);
#else
      *dst = *(const uint4*)src;
#endif
    }
#if HAVE_ASYNC_LDS
#if __has_builtin(__builtin_amdgcn_s_wait_asynccnt)
    __builtin_amdgcn_s_wait_asynccnt(0);
#else
    asm volatile("s_wait_asynccnt 0x0" ::: "memory");
#endif
#endif
    __syncthreads();

    // ---- B fragment (32x16): lane = N, half hi holds 16 contiguous K ------
    FragBF b;
    if constexpr (PREBF) {
      const unsigned short* bp = bb + k0 + hi * 16;
      __builtin_prefetch(bp + 256, 0, 1);
      b.q[0] = ((const uint4*)bp)[0];
      b.q[1] = ((const uint4*)bp)[1];
    } else {
      const float* bp = bf + k0 + hi * 16;
      __builtin_prefetch(bp + 128, 0, 1);
      const float4 f0 = ((const float4*)bp)[0];
      const float4 f1 = ((const float4*)bp)[1];
      const float4 f2 = ((const float4*)bp)[2];
      const float4 f3 = ((const float4*)bp)[3];
      b.u[0]  = f2b(f0.x); b.u[1]  = f2b(f0.y); b.u[2]  = f2b(f0.z); b.u[3]  = f2b(f0.w);
      b.u[4]  = f2b(f1.x); b.u[5]  = f2b(f1.y); b.u[6]  = f2b(f1.z); b.u[7]  = f2b(f1.w);
      b.u[8]  = f2b(f2.x); b.u[9]  = f2b(f2.y); b.u[10] = f2b(f2.z); b.u[11] = f2b(f2.w);
      b.u[12] = f2b(f3.x); b.u[13] = f2b(f3.y); b.u[14] = f2b(f3.z); b.u[15] = f2b(f3.w);
    }

    // ---- 8 WMMAs: full M=128 against this B fragment ----------------------
#pragma unroll
    for (int mt = 0; mt < 8; ++mt) {
      FragBF a;
      a.q[0] = ldsA[(mt * 32 + lane) * 2 + 0];
      a.q[1] = ldsA[(mt * 32 + lane) * 2 + 1];
      acc[mt] = __builtin_amdgcn_wmma_f32_16x16x32_bf16(
          false, a.v, false, b.v, (short)0, acc[mt], false, false);
    }
  }

  // ---- epilogue: D element i -> row = mt*16 + 8*hi + i, col = col0+mr -----
  const int col = col0 + mr;
  const float bc = bias ? bias[col] : 0.0f;
#pragma unroll
  for (int mt = 0; mt < 8; ++mt) {
#pragma unroll
    for (int i = 0; i < 8; ++i) {
      const int r = mt * 16 + (hi << 3) + i;
      float vv = acc[mt][i] + bc;
      if (extra) vv += extra[(size_t)r * lde + col];
      C[(size_t)r * ldc + col] = vv;
      if (Cbf) Cbf[(size_t)r * ldcb + col] = f2b(vv);
    }
  }
}

// ---- K-split partial reduction ----------------------------------------------
__global__ __launch_bounds__(256)
void reduce_partials(const float* __restrict__ part, long pstride,
                     float* __restrict__ C, unsigned short* __restrict__ Cbf, int n)
{
  const int idx = blockIdx.x * 256 + threadIdx.x;
  if (idx >= n) return;
  float s = 0.0f;
#pragma unroll
  for (int p = 0; p < KSPLIT; ++p) s += part[(long)p * pstride + idx];
  C[idx] = s;
  Cbf[idx] = f2b(s);
}

// ---- elementwise GRU cell ---------------------------------------------------
__global__ __launch_bounds__(256)
void gru_cell(const float* __restrict__ gi, const float* __restrict__ gh,
              float* __restrict__ h, unsigned short* __restrict__ hbf)
{
  const int idx = blockIdx.x * 256 + threadIdx.x;  // over B*H
  const int b = idx >> 9;          // H == 512
  const int j = idx & 511;
  const float* gib = gi + b * H3_;
  const float* ghb = gh + b * H3_;
  const float ir = gib[j], iz = gib[H_ + j], in_ = gib[2 * H_ + j];
  const float hr = ghb[j], hz = ghb[H_ + j], hn  = ghb[2 * H_ + j];
  const float r = 1.0f / (1.0f + __expf(-(ir + hr)));
  const float z = 1.0f / (1.0f + __expf(-(iz + hz)));
  const float n = tanhf(in_ + r * hn);
  const float hnew = (1.0f - z) * n + z * h[idx];
  h[idx]   = hnew;
  hbf[idx] = f2b(hnew);
}

// ---- softmax over V=32000 per row ------------------------------------------
__global__ __launch_bounds__(256)
void softmax_stats(const float* __restrict__ X, long ldr,
                   float* __restrict__ rowmax, float* __restrict__ rowsum)
{
  __shared__ float red[256];
  const int b = blockIdx.x, tid = threadIdx.x;
  const float* row = X + (size_t)b * ldr;
  float m = -3.0e38f;
  for (int v = tid; v < V_; v += 256) m = fmaxf(m, row[v]);
  red[tid] = m; __syncthreads();
  for (int s = 128; s > 0; s >>= 1) {
    if (tid < s) red[tid] = fmaxf(red[tid], red[tid + s]);
    __syncthreads();
  }
  m = red[0]; __syncthreads();
  float sum = 0.0f;
  for (int v = tid; v < V_; v += 256) sum += __expf(row[v] - m);
  red[tid] = sum; __syncthreads();
  for (int s = 128; s > 0; s >>= 1) {
    if (tid < s) red[tid] += red[tid + s];
    __syncthreads();
  }
  if (tid == 0) { rowmax[b] = m; rowsum[b] = red[0]; }
}

__global__ __launch_bounds__(256)
void softmax_norm(float* __restrict__ X, long ldr,
                  const float* __restrict__ rowmax, const float* __restrict__ rowsum,
                  unsigned short* __restrict__ xs_bf)
{
  const int idx = blockIdx.x * 256 + threadIdx.x;  // over B*V (exact multiple)
  const int b = idx / V_;
  const int v = idx - b * V_;
  float* p = X + (size_t)b * ldr + v;
  const float val = __expf(*p - rowmax[b]) / rowsum[b];
  *p = val;
  xs_bf[(size_t)b * V_ + v] = f2b(val);
}

// ---- small helpers ----------------------------------------------------------
__global__ __launch_bounds__(256)
void cvt_f32_bf16(const float* __restrict__ in, unsigned short* __restrict__ out, int n)
{
  const int idx = blockIdx.x * 256 + threadIdx.x;
  if (idx < n) out[idx] = f2b(in[idx]);
}

__global__ __launch_bounds__(256)
void bcast_sos(const float* __restrict__ sos, unsigned short* __restrict__ e_bf)
{
  const int idx = blockIdx.x * 256 + threadIdx.x;  // over B*E
  e_bf[idx] = f2b(sos[idx & (E_ - 1)]);
}

// ---- host launcher ----------------------------------------------------------
extern "C" void kernel_launch(void* const* d_in, const int* in_sizes, int n_in,
                              void* d_out, int out_size, void* d_ws, size_t ws_size,
                              hipStream_t stream)
{
  const float* x       = (const float*)d_in[0];
  const float* agent_W = (const float*)d_in[1];
  const float* agent_b = (const float*)d_in[2];
  const float* Wih     = (const float*)d_in[3];
  const float* Whh     = (const float*)d_in[4];
  const float* bih     = (const float*)d_in[5];
  const float* bhh     = (const float*)d_in[6];
  const float* out_W   = (const float*)d_in[7];
  const float* out_b   = (const float*)d_in[8];
  const float* emb_W   = (const float*)d_in[9];
  const float* sos     = (const float*)d_in[10];
  const float* gumbel  = (const float*)d_in[11];
  float* out = (float*)d_out;   // [B, T, V] fp32

  char* ws = (char*)d_ws;
  size_t off = 0;
  auto carve = [&](size_t bytes) -> char* {
    char* p = ws + off;
    off = (off + bytes + 255) & ~(size_t)255;
    return p;
  };

  // activation / scratch buffers (~16 MB)
  unsigned short* x_bf  = (unsigned short*)carve((size_t)B_ * IN_ * 2);
  unsigned short* h_bf  = (unsigned short*)carve((size_t)B_ * H_ * 2);
  unsigned short* e_bf  = (unsigned short*)carve((size_t)B_ * E_ * 2);
  unsigned short* xs_bf = (unsigned short*)carve((size_t)B_ * V_ * 2);
  float* h_f   = (float*)carve((size_t)B_ * H_ * 4);
  float* e_f   = (float*)carve((size_t)B_ * E_ * 4);
  float* gi    = (float*)carve((size_t)B_ * H3_ * 4);
  float* gh    = (float*)carve((size_t)B_ * H3_ * 4);
  float* part  = (float*)carve((size_t)KSPLIT * B_ * E_ * 4);
  float* rowmx = (float*)carve((size_t)B_ * 4);
  float* rowsm = (float*)carve((size_t)B_ * 4);

  // bf16 weight copies (66 MB) — only if workspace allows
  const size_t nAg = (size_t)H_ * IN_, nIh = (size_t)H3_ * E_, nHh = (size_t)H3_ * H_;
  const size_t nOw = (size_t)V_ * H_,  nEw = (size_t)E_ * V_;
  const size_t wbytes = 2 * (nAg + nIh + nHh + nOw + nEw) + 5 * 256;
  const bool prebf = (ws_size >= off + wbytes);

  unsigned short *agW_bf = nullptr, *Wih_bf = nullptr, *Whh_bf = nullptr,
                 *outW_bf = nullptr, *embW_bf = nullptr;
  if (prebf) {
    agW_bf  = (unsigned short*)carve(nAg * 2);
    Wih_bf  = (unsigned short*)carve(nIh * 2);
    Whh_bf  = (unsigned short*)carve(nHh * 2);
    outW_bf = (unsigned short*)carve(nOw * 2);
    embW_bf = (unsigned short*)carve(nEw * 2);
  }

  const dim3 blk(256);
  auto cvt = [&](const float* src, unsigned short* dst, size_t n) {
    cvt_f32_bf16<<<(unsigned)((n + 255) / 256), blk, 0, stream>>>(src, dst, (int)n);
  };

  if (prebf) {
    cvt(agent_W, agW_bf, nAg);
    cvt(Wih, Wih_bf, nIh);
    cvt(Whh, Whh_bf, nHh);
    cvt(out_W, outW_bf, nOw);
    cvt(emb_W, embW_bf, nEw);
  }

  // unified GEMM launcher: picks pre-bf16 or fp32-convert path
  auto gemm = [&](const unsigned short* A, int lda,
                  const unsigned short* Bbf, const float* Bf32, int ldb,
                  const float* bias, const float* extra, int lde,
                  float* C, long ldc, long zstride,
                  unsigned short* Cbf, int ldcb,
                  int N, int k_len, int zsplit) {
    dim3 grid(N / 128, 1, zsplit);
    if (prebf)
      gemm_core<true><<<grid, blk, 0, stream>>>(A, lda, (const void*)Bbf, ldb,
          bias, extra, lde, C, ldc, zstride, Cbf, ldcb, k_len);
    else
      gemm_core<false><<<grid, blk, 0, stream>>>(A, lda, (const void*)Bf32, ldb,
          bias, extra, lde, C, ldc, zstride, Cbf, ldcb, k_len);
  };

  // h0 = x @ agent_W.T + agent_b ; e0 = broadcast(sos)
  cvt(x, x_bf, (size_t)B_ * IN_);
  gemm(x_bf, IN_, agW_bf, agent_W, IN_, agent_b, nullptr, 0,
       h_f, H_, 0, h_bf, H_, H_, IN_, 1);
  bcast_sos<<<(B_ * E_) / 256, blk, 0, stream>>>(sos, e_bf);

  for (int t = 0; t < T_; ++t) {
    // gi = e @ Wih.T + bih ; gh = h @ Whh.T + bhh
    gemm(e_bf, E_, Wih_bf, Wih, E_, bih, nullptr, 0,
         gi, H3_, 0, nullptr, 0, H3_, E_, 1);
    gemm(h_bf, H_, Whh_bf, Whh, H_, bhh, nullptr, 0,
         gh, H3_, 0, nullptr, 0, H3_, H_, 1);
    gru_cell<<<(B_ * H_) / 256, blk, 0, stream>>>(gi, gh, h_f, h_bf);

    // logits + gumbel[t] -> d_out[b, t, :]
    gemm(h_bf, H_, outW_bf, out_W, H_, out_b,
         gumbel + (size_t)t * B_ * V_, V_,
         out + (size_t)t * V_, (long)T_ * V_, 0, nullptr, 0, V_, H_, 1);

    // softmax over V in place; emit xs in bf16 for the embedding GEMM
    softmax_stats<<<B_, blk, 0, stream>>>(out + (size_t)t * V_, (long)T_ * V_, rowmx, rowsm);
    softmax_norm<<<(B_ * V_) / 256, blk, 0, stream>>>(
        out + (size_t)t * V_, (long)T_ * V_, rowmx, rowsm, xs_bf);

    // e_t = xs @ emb_W.T  — K=32000 split across gridDim.z for parallelism
    gemm(xs_bf, V_, embW_bf, emb_W, V_, nullptr, nullptr, 0,
         part, E_, (long)B_ * E_, nullptr, 0, E_, V_ / KSPLIT, KSPLIT);
    reduce_partials<<<(B_ * E_) / 256, blk, 0, stream>>>(
        part, (long)B_ * E_, e_f, e_bf, B_ * E_);
  }
}
